// CausalSelfAttention_38714835206799
// MI455X (gfx1250) — matrix operand order repaired and checked by reading, not compiled, not run on hardware
//
#include <hip/hip_runtime.h>

// Problem constants (from reference): T=4096, C=1024, H=16, D=64
#define TT 4096
#define CC 1024
#define HH 16
#define DD 64

typedef __bf16 bf16;
typedef __attribute__((ext_vector_type(16))) __bf16 v16bf;
typedef __attribute__((ext_vector_type(8)))  float  v8f;

union Frag { v16bf v; uint4 q[2]; };

__device__ __forceinline__ unsigned short f2bf_bits(float f) {
  unsigned u = __builtin_bit_cast(unsigned, f);
  u += 0x7FFFu + ((u >> 16) & 1u);   // round-to-nearest-even
  return (unsigned short)(u >> 16);
}
__device__ __forceinline__ bf16 f2bf(float f) {
  return __builtin_bit_cast(bf16, f2bf_bits(f));
}

// ---------------------------------------------------------------- converts
__global__ void cvt_f32_bf16(const float* __restrict__ in, bf16* __restrict__ out, int n) {
  int i = blockIdx.x * blockDim.x + threadIdx.x;
  int stride = gridDim.x * blockDim.x;
  for (; i < n; i += stride) out[i] = f2bf(in[i]);
}

// out[c*R + r] = bf16(in[r*Ccols + c])   (B^T so GEMM B-frags load contiguously)
__global__ void transpose_f32_bf16(const float* __restrict__ in, bf16* __restrict__ out,
                                   int R, int Ccols) {
  int i = blockIdx.x * blockDim.x + threadIdx.x;
  int n = R * Ccols;
  int stride = gridDim.x * blockDim.x;
  for (; i < n; i += stride) {
    int r = i / Ccols, c = i - r * Ccols;
    out[(size_t)c * R + r] = f2bf(in[i]);
  }
}

// qkv (T x 3C, bf16 row-major) -> q[h][t][d], k[h][t][d], vT[h][d][t]
__global__ void pack_qkv(const bf16* __restrict__ qkv, bf16* __restrict__ q,
                         bf16* __restrict__ k, bf16* __restrict__ vt) {
  int i = blockIdx.x * blockDim.x + threadIdx.x;
  const int n = TT * 3 * CC;
  int stride = gridDim.x * blockDim.x;
  for (; i < n; i += stride) {
    int t = i / (3 * CC);
    int c = i - t * (3 * CC);
    bf16 val = qkv[i];
    if (c < CC) {
      int h = c >> 6, d = c & 63;
      q[((size_t)h * TT + t) * DD + d] = val;
    } else if (c < 2 * CC) {
      int c2 = c - CC; int h = c2 >> 6, d = c2 & 63;
      k[((size_t)h * TT + t) * DD + d] = val;
    } else {
      int c2 = c - 2 * CC; int h = c2 >> 6, d = c2 & 63;
      vt[((size_t)h * DD + d) * TT + t] = val;   // transposed V
    }
  }
}

// ---------------------------------------------------------------- GEMM
// C(MxN) = A(MxK) * B(KxN);  A row-major bf16, BT = B^T (N x K) row-major bf16.
// Block: 256 threads = 8 waves; wave tile 64x64 (4x4 wmma tiles); block tile 256x128.
template <bool OUT_BF16>
__global__ __launch_bounds__(256)
void gemm_bf16_wmma(const bf16* __restrict__ A, const bf16* __restrict__ BT,
                    void* __restrict__ Cout, int M, int N, int K) {
  const int lane = threadIdx.x & 31;
  const int wave = threadIdx.x >> 5;
  const int l16  = lane & 15;
  const int hs   = lane >> 4;          // 0: lanes 0-15, 1: lanes 16-31
  const int wm   = blockIdx.y * 256 + (wave >> 1) * 64;   // 4 waves in M
  const int wn   = blockIdx.x * 128 + (wave & 1) * 64;    // 2 waves in N

  v8f acc[4][4];
#pragma unroll
  for (int mt = 0; mt < 4; ++mt)
#pragma unroll
    for (int nt = 0; nt < 4; ++nt)
#pragma unroll
      for (int i = 0; i < 8; ++i) acc[mt][nt][i] = 0.0f;

  for (int k0 = 0; k0 < K; k0 += 32) {
    Frag a[4];
#pragma unroll
    for (int mt = 0; mt < 4; ++mt) {
      // A layout: row M=l16; half0 K {0..7,16..23}, half1 {8..15,24..31}
      const bf16* ap = A + (size_t)(wm + mt * 16 + l16) * K + k0 + hs * 8;
      a[mt].q[0] = *(const uint4*)(ap);
      a[mt].q[1] = *(const uint4*)(ap + 16);
    }
    Frag b[4];
#pragma unroll
    for (int nt = 0; nt < 4; ++nt) {
      // B layout: col N=l16; half0 K {0..7,16..23}, half1 {8..15,24..31}
      const bf16* bp = BT + (size_t)(wn + nt * 16 + l16) * K + k0 + hs * 8;
      b[nt].q[0] = *(const uint4*)(bp);
      b[nt].q[1] = *(const uint4*)(bp + 16);
    }
#pragma unroll
    for (int mt = 0; mt < 4; ++mt)
#pragma unroll
      for (int nt = 0; nt < 4; ++nt)
        acc[mt][nt] = __builtin_amdgcn_wmma_f32_16x16x32_bf16(
            false, a[mt].v, false, b[nt].v, (short)0, acc[mt][nt], false, false);
  }

#pragma unroll
  for (int mt = 0; mt < 4; ++mt)
#pragma unroll
    for (int nt = 0; nt < 4; ++nt)
#pragma unroll
      for (int i = 0; i < 8; ++i) {
        int row = wm + mt * 16 + hs * 8 + i;   // C layout: reg i -> row i / i+8
        int col = wn + nt * 16 + l16;
        float v = acc[mt][nt][i];
        if (OUT_BF16) ((bf16*)Cout)[(size_t)row * N + col] = f2bf(v);
        else          ((float*)Cout)[(size_t)row * N + col] = v;
      }
}

// ---------------------------------------------------------------- flash attention
// One wave per (head, 32-query tile): two 16-row M-tiles share every K/V fragment.
// Key blocks of 32. Online softmax in f32.
__global__ __launch_bounds__(32)
void flash_attn(const bf16* __restrict__ Q,   // [H][T][D]
                const bf16* __restrict__ Kd,  // [H][T][D]
                const bf16* __restrict__ Vt,  // [H][D][T]
                bf16* __restrict__ Y,         // [T][C]
                const int* __restrict__ nf_ptr) {
  const int lane = threadIdx.x & 31;
  const int l16  = lane & 15;
  const int hs   = lane >> 4;
  const int qt   = blockIdx.x;
  const int h    = blockIdx.y;
  const int nf   = *nf_ptr;
  const int q0   = qt * 32;

  __shared__ __align__(16) unsigned short pTile[32 * 32];

  // Q fragments: two 16x64 tiles -> 2 x (two 16x32 A-frags)
  Frag qa[2][2];
#pragma unroll
  for (int mt = 0; mt < 2; ++mt) {
    const bf16* qp = Q + ((size_t)h * TT + q0 + mt * 16 + l16) * DD + hs * 8;
    qa[mt][0].q[0] = *(const uint4*)(qp);
    qa[mt][0].q[1] = *(const uint4*)(qp + 16);
    qa[mt][1].q[0] = *(const uint4*)(qp + 32);
    qa[mt][1].q[1] = *(const uint4*)(qp + 48);
  }

  v8f o[2][4];
  float mrow[2][8], lrow[2][8];
#pragma unroll
  for (int mt = 0; mt < 2; ++mt) {
#pragma unroll
    for (int nt = 0; nt < 4; ++nt)
#pragma unroll
      for (int i = 0; i < 8; ++i) o[mt][nt][i] = 0.0f;
#pragma unroll
    for (int i = 0; i < 8; ++i) { mrow[mt][i] = -1e30f; lrow[mt][i] = 0.0f; }
  }

  int SEnd = q0 + 32;
  if (nf > SEnd) SEnd = nf;
  if (SEnd > TT) SEnd = TT;

  for (int s0 = 0; s0 < SEnd; s0 += 32) {
    // ---- K fragments for this 32-key block (shared by both M-tiles)
    Frag kb[2][2];   // [ntile][chunk]
#pragma unroll
    for (int nt = 0; nt < 2; ++nt) {
      const bf16* kp = Kd + ((size_t)h * TT + s0 + nt * 16 + l16) * DD + hs * 8;
      kb[nt][0].q[0] = *(const uint4*)(kp);
      kb[nt][0].q[1] = *(const uint4*)(kp + 16);
      kb[nt][1].q[0] = *(const uint4*)(kp + 32);
      kb[nt][1].q[1] = *(const uint4*)(kp + 48);
    }

    // ---- scores S = Q * K^T  (2 M-tiles x 2 N-tiles, K-dim = 64 in two chunks)
    v8f s[2][2];
#pragma unroll
    for (int mt = 0; mt < 2; ++mt)
#pragma unroll
      for (int nt = 0; nt < 2; ++nt) {
        v8f z;
#pragma unroll
        for (int i = 0; i < 8; ++i) z[i] = 0.0f;
        z = __builtin_amdgcn_wmma_f32_16x16x32_bf16(false, qa[mt][0].v, false, kb[nt][0].v,
                                                    (short)0, z, false, false);
        s[mt][nt] = __builtin_amdgcn_wmma_f32_16x16x32_bf16(false, qa[mt][1].v, false, kb[nt][1].v,
                                                            (short)0, z, false, false);
      }

    // ---- scale + mask (causal OR global prefix), online softmax per M-tile
    float sv[2][2][8];
#pragma unroll
    for (int mt = 0; mt < 2; ++mt)
#pragma unroll
      for (int nt = 0; nt < 2; ++nt)
#pragma unroll
        for (int i = 0; i < 8; ++i) {
          float x = s[mt][nt][i] * 0.125f;        // 1/sqrt(64)
          int row = q0 + mt * 16 + hs * 8 + i;
          int col = s0 + nt * 16 + l16;
          if (col > row && col >= nf) x = -1e30f;
          sv[mt][nt][i] = x;
        }

#pragma unroll
    for (int mt = 0; mt < 2; ++mt) {
      float mnew[8];
#pragma unroll
      for (int i = 0; i < 8; ++i) {
        float t = fmaxf(sv[mt][0][i], sv[mt][1][i]);
        t = fmaxf(t, __shfl_xor(t, 1, 32));
        t = fmaxf(t, __shfl_xor(t, 2, 32));
        t = fmaxf(t, __shfl_xor(t, 4, 32));
        t = fmaxf(t, __shfl_xor(t, 8, 32));
        mnew[i] = fmaxf(mrow[mt][i], t);
      }
#pragma unroll
      for (int i = 0; i < 8; ++i) {
        float alpha = __expf(mrow[mt][i] - mnew[i]);
        mrow[mt][i] = mnew[i];
        float p0 = __expf(sv[mt][0][i] - mnew[i]);
        float p1 = __expf(sv[mt][1][i] - mnew[i]);
        sv[mt][0][i] = p0; sv[mt][1][i] = p1;
        float rs = p0 + p1;
        rs += __shfl_xor(rs, 1, 32);
        rs += __shfl_xor(rs, 2, 32);
        rs += __shfl_xor(rs, 4, 32);
        rs += __shfl_xor(rs, 8, 32);
        lrow[mt][i] = lrow[mt][i] * alpha + rs;
#pragma unroll
        for (int nt = 0; nt < 4; ++nt) o[mt][nt][i] *= alpha;
      }
    }

    // ---- C-layout -> A-layout via LDS (P as bf16, 32x32 tile)
    __syncthreads();
#pragma unroll
    for (int mt = 0; mt < 2; ++mt)
#pragma unroll
      for (int i = 0; i < 8; ++i) {
        int r = mt * 16 + hs * 8 + i;
        pTile[r * 32 + l16]      = f2bf_bits(sv[mt][0][i]);
        pTile[r * 32 + 16 + l16] = f2bf_bits(sv[mt][1][i]);
      }
    __syncthreads();
    Frag pf[2];
#pragma unroll
    for (int mt = 0; mt < 2; ++mt) {
      const unsigned short* pp = &pTile[(mt * 16 + l16) * 32 + hs * 8];
      pf[mt].q[0] = *(const uint4*)(pp);
      pf[mt].q[1] = *(const uint4*)(pp + 16);
    }

    // ---- O += P * V   (V transposed: contiguous B-frags, shared by both M-tiles)
#pragma unroll
    for (int nt = 0; nt < 4; ++nt) {
      Frag vf;
      const bf16* vp = Vt + ((size_t)h * DD + nt * 16 + l16) * TT + s0 + hs * 8;
      vf.q[0] = *(const uint4*)(vp);
      vf.q[1] = *(const uint4*)(vp + 16);
#pragma unroll
      for (int mt = 0; mt < 2; ++mt)
        o[mt][nt] = __builtin_amdgcn_wmma_f32_16x16x32_bf16(false, pf[mt].v, false, vf.v,
                                                            (short)0, o[mt][nt], false, false);
    }
  }

  // ---- normalize and store y (bf16, [T][C] layout for the projection GEMM)
#pragma unroll
  for (int mt = 0; mt < 2; ++mt)
#pragma unroll
    for (int i = 0; i < 8; ++i) {
      float inv = 1.0f / lrow[mt][i];
      int row = q0 + mt * 16 + hs * 8 + i;
#pragma unroll
      for (int nt = 0; nt < 4; ++nt) {
        int col = h * DD + nt * 16 + l16;
        Y[(size_t)row * CC + col] = f2bf(o[mt][nt][i] * inv);
      }
    }
}

// ---------------------------------------------------------------- launch
extern "C" void kernel_launch(void* const* d_in, const int* in_sizes, int n_in,
                              void* d_out, int out_size, void* d_ws, size_t ws_size,
                              hipStream_t stream) {
  const float* x      = (const float*)d_in[0];   // (T, C)
  const float* w_attn = (const float*)d_in[1];   // (C, 3C)
  const float* w_proj = (const float*)d_in[2];   // (C, C)
  const int*   nf     = (const int*)d_in[3];     // scalar
  float* out = (float*)d_out;                    // (T, C)

  char* ws = (char*)d_ws;
  size_t off = 0;
  auto alloc = [&](size_t bytes) { char* p = ws + off; off = (off + bytes + 255) & ~(size_t)255; return p; };

  bf16* x_bf   = (bf16*)alloc((size_t)TT * CC * 2);
  bf16* wattnT = (bf16*)alloc((size_t)3 * CC * CC * 2);   // (3C, C)
  bf16* wprojT = (bf16*)alloc((size_t)CC * CC * 2);       // (C, C)
  bf16* qkv    = (bf16*)alloc((size_t)TT * 3 * CC * 2);   // (T, 3C)
  bf16* qb     = (bf16*)alloc((size_t)HH * TT * DD * 2);
  bf16* kb     = (bf16*)alloc((size_t)HH * TT * DD * 2);
  bf16* vtb    = (bf16*)alloc((size_t)HH * DD * TT * 2);
  bf16* yb     = (bf16*)alloc((size_t)TT * CC * 2);

  // 1. convert / transpose weights+activations to bf16
  {
    int n = TT * CC;
    cvt_f32_bf16<<<(n + 255) / 256, 256, 0, stream>>>(x, x_bf, n);
    transpose_f32_bf16<<<(CC * 3 * CC + 255) / 256, 256, 0, stream>>>(w_attn, wattnT, CC, 3 * CC);
    transpose_f32_bf16<<<(CC * CC + 255) / 256, 256, 0, stream>>>(w_proj, wprojT, CC, CC);
  }

  // 2. qkv = x @ w_attn  (M=4096, N=3072, K=1024), bf16 out
  gemm_bf16_wmma<true><<<dim3(3 * CC / 128, TT / 256), 256, 0, stream>>>(
      x_bf, wattnT, qkv, TT, 3 * CC, CC);

  // 3. split heads; transpose V
  pack_qkv<<<(TT * 3 * CC + 255) / 256, 256, 0, stream>>>(qkv, qb, kb, vtb);

  // 4. flash attention: one wave per (head, 32-row query tile)
  flash_attn<<<dim3(TT / 32, HH), 32, 0, stream>>>(qb, kb, vtb, yb, nf);

  // 5. out = y @ w_proj  (M=4096, N=1024, K=1024), f32 out
  gemm_bf16_wmma<false><<<dim3(CC / 128, TT / 256), 256, 0, stream>>>(
      yb, wprojT, out, TT, CC, CC);
}
